// RnnLayer_9852654977437
// MI455X (gfx1250) — matrix operand order repaired and checked by reading, not compiled, not run on hardware
//
#include <hip/hip_runtime.h>
#include <math.h>

// Shapes (fixed by the reference): B=64, T=512, D=768, H=384, 4H=1536.
#define BT_ROWS   32768      // B*T
#define T_        512
#define D_        768
#define H_        384
#define G4H_      1536
#define NT_TOT    96         // 1536/16 N-tiles
#define KT_WX     24         // 768/32 K-tiles
#define KT_WH     12         // 384/32 K-tiles
#define MT_X      2048       // 32768/16 M-tiles of x

typedef __attribute__((ext_vector_type(16))) __bf16  bf16x16;
typedef __attribute__((ext_vector_type(8)))  float   v8f;

union Frag { unsigned u[8]; bf16x16 v; };

__device__ __forceinline__ unsigned f2bf(float f) {
  unsigned x = __float_as_uint(f);
  return (x + 0x7FFFu + ((x >> 16) & 1u)) >> 16;   // RNE float->bf16
}

__device__ __forceinline__ v8f wmma_bf16(bf16x16 a, bf16x16 b, v8f c) {
  // (neg_a, A, neg_b, B, c_mod, C, reuse_a, reuse_b)
  return __builtin_amdgcn_wmma_f32_16x16x32_bf16(false, a, false, b, (short)0, c,
                                                 false, false);
}

__device__ __forceinline__ float sigmoid_fast(float x) {
  return 1.0f / (1.0f + __expf(-x));
}

__device__ __forceinline__ void load_frag(const unsigned* __restrict__ base,
                                          size_t fragIdx, int lane, Frag& f) {
  const uint4* p = reinterpret_cast<const uint4*>(base + ((fragIdx * 32 + lane) << 3));
  uint4 a = p[0], b = p[1];
  f.u[0] = a.x; f.u[1] = a.y; f.u[2] = a.z; f.u[3] = a.w;
  f.u[4] = b.x; f.u[5] = b.y; f.u[6] = b.z; f.u[7] = b.w;
}

// ---------------------------------------------------------------------------
// Pack a (KT*32) x 1536 f32 weight matrix into bf16 WMMA B-fragment order:
// frag(nt,kt): lane l, dword v holds W[k]|W[k+1], k = kt*32 + {2v | 2v+8 for v>=4} + 8*(l>=16),
// n = nt*16 + (l&15). One thread per output dword; thread id == output index.
// ---------------------------------------------------------------------------
__global__ void pack_w_kernel(const float* __restrict__ Wsrc, int KT,
                              unsigned* __restrict__ outp) {
  int t = blockIdx.x * blockDim.x + threadIdx.x;
  int v = t & 7;
  int l = (t >> 3) & 31;
  int f = t >> 8;
  if (f >= NT_TOT * KT) return;
  int nt = f / KT, kt = f - nt * KT;
  int k = kt * 32 + ((v < 4) ? 2 * v : 2 * v + 8) + ((l & 16) ? 8 : 0);
  int n = nt * 16 + (l & 15);
  float f0 = Wsrc[(size_t)k * G4H_ + n];
  float f1 = Wsrc[(size_t)(k + 1) * G4H_ + n];
  outp[t] = f2bf(f0) | (f2bf(f1) << 16);
}

// ---------------------------------------------------------------------------
// Pack x (BT x 768 f32) into bf16 WMMA 16-bit A-fragment order:
// frag(mt,kt): lane l, dword v holds x[m][k]|x[m][k+1] with m = mt*16+(l&15),
// k = kt*32 + {2v | 2v+8 for v>=4} + 8*(l>=16). One thread per output dword.
// ---------------------------------------------------------------------------
__global__ void pack_x_kernel(const float* __restrict__ x,
                              unsigned* __restrict__ outp) {
  int t = blockIdx.x * blockDim.x + threadIdx.x;
  int v  = t & 7;
  int l  = (t >> 3) & 31;
  int f  = t >> 8;
  int mt = f / KT_WX, kt = f - mt * KT_WX;
  int m = mt * 16 + (l & 15);
  int k = kt * 32 + ((v < 4) ? 2 * v : 2 * v + 8) + ((l & 16) ? 8 : 0);
  float2 xv = *reinterpret_cast<const float2*>(x + (size_t)m * D_ + k);
  outp[t] = f2bf(xv.x) | (f2bf(xv.y) << 16);
}

// ---------------------------------------------------------------------------
// Phase 1: xg[m, n] = x[m, :] @ Wx[:, n] + b[n], m = b*T + t.
// One wave computes a 16x128 output tile (8 WMMA n-tiles), K-loop of 24.
// A/B fragments are pre-packed bf16, software-pipelined one iteration ahead.
// The (kt,nt) loops are FULLY unrolled so the double-buffer indices are
// compile-time constants -> fragments stay in VGPRs (no alloca->LDS bounce).
// grid = (BT/128, 1536/128), block = 256 (8 waves).
// ---------------------------------------------------------------------------
__global__ void xg_gemm_kernel(const unsigned* __restrict__ packA,
                               const unsigned* __restrict__ packB,
                               const float* __restrict__ bias,
                               float* __restrict__ xg) {
  const int lane = threadIdx.x & 31;
  const int wv   = threadIdx.x >> 5;
  const int l15  = lane & 15;
  const int hi8  = (lane & 16) ? 8 : 0;
  const int mt   = blockIdx.x * 8 + wv;
  const int m0   = mt * 16;

  v8f acc[8];
#pragma unroll
  for (int i = 0; i < 8; ++i) acc[i] = {};

  const size_t aOff = (size_t)mt * KT_WX;                 // frag idx of (mt, kt=0)
  const size_t bOff = (size_t)(blockIdx.y * 8) * KT_WX;   // frag idx of (ntG0, kt=0)

  Frag a2[2], b2[2];
  load_frag(packA, aOff, lane, a2[0]);
  load_frag(packB, bOff, lane, b2[0]);

#pragma unroll
  for (int kt = 0; kt < KT_WX; ++kt) {
#pragma unroll
    for (int nt = 0; nt < 8; ++nt) {
      int it  = kt * 8 + nt;
      int itn = (it + 1 > KT_WX * 8 - 1) ? (KT_WX * 8 - 1) : it + 1;
      int ktn = itn >> 3, ntn = itn & 7;
      if (ntn == 0)   // entering a new k-tile: prefetch its A fragment
        load_frag(packA, aOff + ktn, lane, a2[ktn & 1]);
      load_frag(packB, bOff + (size_t)ntn * KT_WX + ktn, lane, b2[itn & 1]);
      acc[nt] = wmma_bf16(a2[kt & 1].v, b2[it & 1].v, acc[nt]);
    }
  }

#pragma unroll
  for (int nt = 0; nt < 8; ++nt) {
    int ntG = blockIdx.y * 8 + nt;
    int n = ntG * 16 + l15;
    float bs = bias[n];
#pragma unroll
    for (int r = 0; r < 8; ++r) {
      int mrow = m0 + r + hi8;                 // C layout: lanes>=16 hold M=8..15
      xg[(size_t)mrow * G4H_ + n] = acc[nt][r] + bs;
    }
  }
}

// ---------------------------------------------------------------------------
// Phase 2: persistent recurrence. grid = (4 batch tiles, 2 directions),
// block = 256 (8 waves). Wave w owns n-tiles {g*24 + 3w + p} for gate g,
// p<3, so i/j/f/o for its hidden units stay register-resident.
// h is shared via LDS in bf16, row-major [m][k]; A fragments gathered on read.
// Wh B-fragments are double-buffered one WMMA ahead (static parity: kt*12
// is even, so it&1 == q&1 is compile-time under the unrolled q loop).
// ---------------------------------------------------------------------------
__global__ void __launch_bounds__(256)
lstm_rec_kernel(const float* __restrict__ xg_fw, const float* __restrict__ xg_bw,
                const unsigned* __restrict__ pWh_fw, const unsigned* __restrict__ pWh_bw,
                const float* __restrict__ wci_fw, const float* __restrict__ wcf_fw,
                const float* __restrict__ wco_fw,
                const float* __restrict__ wci_bw, const float* __restrict__ wcf_bw,
                const float* __restrict__ wco_bw,
                const int* __restrict__ lengths, float* __restrict__ out) {
  const int mb   = blockIdx.x;
  const int dir  = blockIdx.y;
  const int m0   = mb * 16;
  const int tid  = threadIdx.x;
  const int lane = tid & 31;
  const int wv   = tid >> 5;
  const int l15  = lane & 15;
  const int hi8  = (lane & 16) ? 8 : 0;

  const float*    xg  = dir ? xg_bw : xg_fw;
  const unsigned* pWh = dir ? pWh_bw : pWh_fw;
  const float*    wci = dir ? wci_bw : wci_fw;
  const float*    wcf = dir ? wcf_bw : wcf_fw;
  const float*    wco = dir ? wco_bw : wco_fw;

  // h as bf16: hwords[m][k/2], row stride 193 words (bank-conflict padding)
  __shared__ unsigned hwords[16 * 193];
  for (int i = tid; i < 16 * 193; i += 256) hwords[i] = 0u;

  int Lr[8];
#pragma unroll
  for (int r = 0; r < 8; ++r) Lr[r] = lengths[m0 + r + hi8];

  float wciR[3], wcfR[3], wcoR[3];
  int   nR[3];
#pragma unroll
  for (int p = 0; p < 3; ++p) {
    int n = 48 * wv + 16 * p + l15;
    nR[p] = n;
    wciR[p] = wci[n]; wcfR[p] = wcf[n]; wcoR[p] = wco[n];
  }

  float c[3][8];
#pragma unroll
  for (int p = 0; p < 3; ++p)
#pragma unroll
    for (int r = 0; r < 8; ++r) c[p][r] = 0.0f;

  __syncthreads();

  for (int t = 0; t < T_; ++t) {
    // Per-batch time index (backward dir reads time-reversed xg rows).
    int rowbase[8];
#pragma unroll
    for (int r = 0; r < 8; ++r) {
      int b  = m0 + r + hi8;
      int Lm = Lr[r];
      int tb = dir ? ((t < Lm) ? (Lm - 1 - t) : t) : t;
      rowbase[r] = (b * T_ + tb) * G4H_;
    }

    // Initialize accumulators with the precomputed input projection.
    v8f acc[4][3];
#pragma unroll
    for (int g = 0; g < 4; ++g)
#pragma unroll
      for (int p = 0; p < 3; ++p) {
        int n = g * H_ + nR[p];
#pragma unroll
        for (int r = 0; r < 8; ++r)
          acc[g][p][r] = xg[rowbase[r] + n];
      }

    // G += h @ Wh via WMMA bf16; A from LDS, B double-buffered from L2.
    Frag b2[2];
    load_frag(pWh, (size_t)(3 * wv) * KT_WH, lane, b2[0]);   // q=0, kt=0
#pragma unroll
    for (int kt = 0; kt < KT_WH; ++kt) {
      Frag a;
#pragma unroll
      for (int v = 0; v < 8; ++v) {
        int k = kt * 32 + ((v < 4) ? 2 * v : 2 * v + 8) + hi8;
        a.u[v] = hwords[l15 * 193 + (k >> 1)];
      }
#pragma unroll
      for (int q = 0; q < 12; ++q) {
        int it  = kt * 12 + q;
        int itn = (it + 1 > KT_WH * 12 - 1) ? (KT_WH * 12 - 1) : it + 1;
        int ktn = itn / 12, qn = itn - ktn * 12;
        int ntn = (qn / 3) * 24 + 3 * wv + (qn % 3);
        load_frag(pWh, (size_t)ntn * KT_WH + ktn, lane, b2[itn & 1]);
        int g = q / 3, p = q - g * 3;
        acc[g][p] = wmma_bf16(a.v, b2[it & 1].v, acc[g][p]);
      }
    }

    __syncthreads();   // all hwords reads done before this step's h writes

    unsigned short* hhalf = reinterpret_cast<unsigned short*>(hwords);
#pragma unroll
    for (int p = 0; p < 3; ++p) {
#pragma unroll
      for (int r = 0; r < 8; ++r) {
        float gi = acc[0][p][r];
        float gj = acc[1][p][r];
        float gf = acc[2][p][r];
        float go = acc[3][p][r];
        float cc = c[p][r];
        float ig = sigmoid_fast(gi + wciR[p] * cc);
        float fg = sigmoid_fast(gf + 1.0f + wcfR[p] * cc);   // FORGET_BIAS
        float cn = fg * cc + ig * tanhf(gj);
        float og = sigmoid_fast(go + wcoR[p] * cn);
        float hn = og * tanhf(cn);
        int Lm = Lr[r];
        if (t < Lm) {                       // masked step: keep c, h, out=0
          c[p][r] = cn;
          int m = r + hi8;
          hhalf[m * 386 + nR[p]] = (unsigned short)f2bf(hn);
          int tout = dir ? (Lm - 1 - t) : t;
          out[(size_t)((m0 + m) * T_ + tout) * 768 + dir * H_ + nR[p]] = hn;
        }
      }
    }
    __syncthreads();   // h visible to every wave before next step's WMMA
  }
}

// ---------------------------------------------------------------------------
extern "C" void kernel_launch(void* const* d_in, const int* in_sizes, int n_in,
                              void* d_out, int out_size, void* d_ws, size_t ws_size,
                              hipStream_t stream) {
  (void)in_sizes; (void)n_in; (void)ws_size;
  const float* x      = (const float*)d_in[0];
  const int*   lens   = (const int*)  d_in[1];
  const float* W_fw   = (const float*)d_in[2];
  const float* b_fw   = (const float*)d_in[3];
  const float* wci_fw = (const float*)d_in[4];
  const float* wcf_fw = (const float*)d_in[5];
  const float* wco_fw = (const float*)d_in[6];
  const float* W_bw   = (const float*)d_in[7];
  const float* b_bw   = (const float*)d_in[8];
  const float* wci_bw = (const float*)d_in[9];
  const float* wcf_bw = (const float*)d_in[10];
  const float* wco_bw = (const float*)d_in[11];
  float* out = (float*)d_out;

  char* ws = (char*)d_ws;
  const size_t XG_BYTES  = (size_t)BT_ROWS * G4H_ * sizeof(float);     // 192 MiB each
  const size_t PWX_BYTES = (size_t)NT_TOT * KT_WX * 32 * 8 * 4;        // 2.25 MiB
  const size_t PWH_BYTES = (size_t)NT_TOT * KT_WH * 32 * 8 * 4;        // 1.125 MiB
  const size_t PX_BYTES  = (size_t)MT_X * KT_WX * 32 * 8 * 4;          // 48 MiB
  size_t off = 0;
  float*    xg_fw  = (float*)(ws + off); off += XG_BYTES;
  float*    xg_bw  = (float*)(ws + off); off += XG_BYTES;
  unsigned* pWx_fw = (unsigned*)(ws + off); off += PWX_BYTES;
  unsigned* pWx_bw = (unsigned*)(ws + off); off += PWX_BYTES;
  unsigned* pWh_fw = (unsigned*)(ws + off); off += PWH_BYTES;
  unsigned* pWh_bw = (unsigned*)(ws + off); off += PWH_BYTES;
  unsigned* pX     = (unsigned*)(ws + off); off += PX_BYTES;

  // Pack weights and x into WMMA fragment order (bf16).
  pack_w_kernel<<<NT_TOT * KT_WX, 256, 0, stream>>>(W_fw, KT_WX, pWx_fw);
  pack_w_kernel<<<NT_TOT * KT_WX, 256, 0, stream>>>(W_bw, KT_WX, pWx_bw);
  pack_w_kernel<<<NT_TOT * KT_WH, 256, 0, stream>>>(W_fw + (size_t)D_ * G4H_, KT_WH, pWh_fw);
  pack_w_kernel<<<NT_TOT * KT_WH, 256, 0, stream>>>(W_bw + (size_t)D_ * G4H_, KT_WH, pWh_bw);
  pack_x_kernel<<<MT_X * KT_WX, 256, 0, stream>>>(x, pX);

  // Phase 1: big input-projection GEMMs (A and B both packed bf16).
  dim3 g1(BT_ROWS / 128, G4H_ / 128);
  xg_gemm_kernel<<<g1, 256, 0, stream>>>(pX, pWx_fw, b_fw, xg_fw);
  xg_gemm_kernel<<<g1, 256, 0, stream>>>(pX, pWx_bw, b_bw, xg_bw);

  // Masked timesteps must read as zero in the output.
  hipMemsetAsync(d_out, 0, (size_t)out_size * sizeof(float), stream);

  // Phase 2: both directions' recurrences run concurrently.
  lstm_rec_kernel<<<dim3(4, 2), 256, 0, stream>>>(
      xg_fw, xg_bw, pWh_fw, pWh_bw,
      wci_fw, wcf_fw, wco_fw, wci_bw, wcf_bw, wco_bw, lens, out);
}
